// Swin_BiFormerBlock_54030688583937
// MI455X (gfx1250) — compile-verified, matrix operand order checked by reading, not run on hardware
//
#include <hip/hip_runtime.h>
#include <math.h>

typedef __attribute__((ext_vector_type(16))) _Float16 v16h;
typedef __attribute__((ext_vector_type(8)))  float    v8f;
typedef _Float16 f16;

#define HH   192
#define WWID 192
#define CC   96
#define BB   2
#define HWN  (HH*WWID)       // 36864
#define NT   (BB*HWN)        // 73728 tokens
#define P2   576             // windows per batch
#define W2   64              // tokens per window
#define NHEAD 8
#define HD   12
#define TOPK 4
#define LKEY (TOPK*W2)       // 256 keys per window
#define MBLK 128             // GEMM block row tile

// ---- WMMA fragment helpers (ISA 7.12.2 16-bit 16x32 layout) ----
// lanes 0-15: rows 0-15, K base 0 ; lanes 16-31: rows 0-15, K base 8
// halves 0..7 -> K = base+0..7 ; halves 8..15 -> K = base+16..23
__device__ __forceinline__ v16h load_frag(const f16* __restrict__ base,
                                          int rs, int cs, int kvalid) {
  const int lane = threadIdx.x & 31;
  const int row  = lane & 15;
  const int kb   = (lane & 16) ? 8 : 0;
  v16h f;
#pragma unroll
  for (int h = 0; h < 16; ++h) {
    int k = kb + (h & 7) + ((h >> 3) << 4);
    f[h] = (k < kvalid) ? base[row * rs + k * cs] : (f16)0.0f;
  }
  return f;
}

__device__ __forceinline__ v8f wmma_f16(v16h a, v16h b, v8f c) {
  return __builtin_amdgcn_wmma_f32_16x16x32_f16(false, a, false, b,
                                                (short)0, c, false, false);
}

// ---- async global->LDS b128 copy of one 16B chunk (CDNA5, ASYNCcnt) ----
__device__ __forceinline__ void async_copy16(unsigned lds_byte_addr,
                                             const void* gptr) {
  unsigned long long ga = (unsigned long long)(uintptr_t)gptr;
  asm volatile("global_load_async_to_lds_b128 %0, %1, off"
               :: "v"(lds_byte_addr), "v"(ga) : "memory");
}
__device__ __forceinline__ void async_wait0() {
  asm volatile("s_wait_asynccnt 0" ::: "memory");
}

// ==== LDS-staged GEMM: C(128 x NOUT) = A(128x96) @ Bw(NOUT x 96)^T ====
// A staged via async-to-LDS, Bw (weights, row-major (N,K)) staged via b128.
// Each wave keeps its 3 A-fragments in registers across all N tiles.
template <int NOUT, typename Epi>
__device__ __forceinline__ void gemm_staged(const f16* __restrict__ A,
                                            const f16* __restrict__ Bw,
                                            Epi epi) {
  constexpr int NTI = NOUT / 16;
  __shared__ alignas(16) f16 sA[MBLK * CC];
  __shared__ alignas(16) f16 sB[NOUT * CC];
  const int tid = threadIdx.x;

  // stage B: NOUT*96 halfs, 16B vector copies
  for (int i = tid * 8; i < NOUT * CC; i += 256 * 8)
    *(uint4*)&sB[i] = *(const uint4*)&Bw[i];

  // stage A: 128*96*2 = 24576 B = 6 x (256 lanes x 16B) async chunks
  {
    unsigned ldsA = (unsigned)(uintptr_t)(void*)&sA[0];
    const char* gA = (const char*)A;
#pragma unroll
    for (int it = 0; it < 6; ++it) {
      unsigned boff = (unsigned)(tid + it * 256) * 16u;
      async_copy16(ldsA + boff, gA + boff);
    }
    async_wait0();
  }
  __syncthreads();

  const int wave = tid >> 5;
  const f16* arow = &sA[(wave * 16) * CC];
  v16h a0 = load_frag(arow + 0,  CC, 1, 32);
  v16h a1 = load_frag(arow + 32, CC, 1, 32);
  v16h a2 = load_frag(arow + 64, CC, 1, 32);
#pragma unroll
  for (int nt = 0; nt < NTI; ++nt) {
    const f16* brow = &sB[(nt * 16) * CC];
    v8f acc = {};
    acc = wmma_f16(a0, load_frag(brow + 0,  CC, 1, 32), acc);
    acc = wmma_f16(a1, load_frag(brow + 32, CC, 1, 32), acc);
    acc = wmma_f16(a2, load_frag(brow + 64, CC, 1, 32), acc);
    epi(wave, nt, acc);
  }
}

// ---- fp32 -> fp16 conversion for weights ----
__global__ void k_cvt16(const float* __restrict__ s, f16* __restrict__ d, int n) {
  int i = blockIdx.x * blockDim.x + threadIdx.x;
  if (i < n) d[i] = (f16)s[i];
}

// ---- LN1 + window partition: x NCHW -> xw fp16 window-token-major ----
__global__ void k_ln1_win(const float* __restrict__ x, const float* __restrict__ g,
                          const float* __restrict__ bta, f16* __restrict__ xw) {
  int m = blockIdx.x * blockDim.x + threadIdx.x;
  if (m >= NT) return;
  int b = m / HWN, t = m % HWN;
  int y = t / WWID, xx = t % WWID;
  const float* xp = x + (size_t)b * CC * HWN + t;
  float buf[CC];
  float mean = 0.f;
#pragma unroll 4
  for (int c = 0; c < CC; ++c) { buf[c] = xp[(size_t)c * HWN]; mean += buf[c]; }
  mean *= (1.0f / CC);
  float var = 0.f;
#pragma unroll 4
  for (int c = 0; c < CC; ++c) { float d = buf[c] - mean; var += d * d; }
  var *= (1.0f / CC);
  float rstd = rsqrtf(var + 1e-5f);
  int p  = (y >> 3) * (WWID >> 3) + (xx >> 3);
  int wi = (y & 7) * 8 + (xx & 7);
  f16* op = xw + ((size_t)(b * P2 + p) * W2 + wi) * CC;
#pragma unroll 4
  for (int c = 0; c < CC; ++c)
    op[c] = (f16)((buf[c] - mean) * rstd * g[c] + bta[c]);
}

// ---- QKV GEMM (staged): (NT x 96) @ (96 x 288) -> q/k/v fp16 ----
__global__ void __launch_bounds__(256)
k_qkv(const f16* __restrict__ xw, const f16* __restrict__ wq,
      const float* __restrict__ bias, f16* __restrict__ q,
      f16* __restrict__ k, f16* __restrict__ v) {
  const size_t row0 = (size_t)blockIdx.x * MBLK;
  gemm_staged<3 * CC>(xw + row0 * CC, wq, [&](int mw, int nt, v8f acc) {
    int lane = threadIdx.x & 31, n0 = lane & 15, mb = (lane & 16) ? 8 : 0;
    int ng = nt * 16 + n0;
    float bv = bias[ng];
    f16* dst; int col;
    if (ng < CC)          { dst = q; col = ng; }
    else if (ng < 2 * CC) { dst = k; col = ng - CC; }
    else                  { dst = v; col = ng - 2 * CC; }
#pragma unroll
    for (int r = 0; r < 8; ++r)
      dst[(row0 + mw * 16 + mb + r) * CC + col] = (f16)(acc[r] + bv);
  });
}

// ---- per-window q/k means for routing ----
__global__ void k_means(const f16* __restrict__ q, const f16* __restrict__ k,
                        f16* __restrict__ qm, f16* __restrict__ km) {
  int i = blockIdx.x * blockDim.x + threadIdx.x;
  if (i >= BB * P2 * CC) return;
  int bp = i / CC, c = i % CC;
  const f16* qp = q + (size_t)bp * W2 * CC + c;
  const f16* kp = k + (size_t)bp * W2 * CC + c;
  float sq = 0.f, sk = 0.f;
  for (int t = 0; t < W2; ++t) { sq += (float)qp[t * CC]; sk += (float)kp[t * CC]; }
  qm[i] = (f16)(sq * (1.0f / W2));
  km[i] = (f16)(sk * (1.0f / W2));
}

// ---- routing logits: per batch 576x576, K=96 (scale irrelevant for top-k) ----
__global__ void k_route(const f16* __restrict__ qm, const f16* __restrict__ km,
                        float* __restrict__ rl) {
  int wave = threadIdx.x >> 5;
  int tile = blockIdx.x * 8 + wave;
  const int T = P2 / 16;                     // 36
  if (tile >= BB * T * T) return;
  int b = tile / (T * T), tl = tile % (T * T);
  int mt = tl / T, nt = tl % T;
  const f16* ab = qm + ((size_t)b * P2 + mt * 16) * CC;
  const f16* bb = km + ((size_t)b * P2 + nt * 16) * CC;
  v8f acc = {};
#pragma unroll
  for (int ks = 0; ks < 3; ++ks)
    acc = wmma_f16(load_frag(ab + ks * 32, CC, 1, 32),
                   load_frag(bb + ks * 32, CC, 1, 32), acc);
  int lane = threadIdx.x & 31, n0 = lane & 15, mb = (lane & 16) ? 8 : 0;
#pragma unroll
  for (int r = 0; r < 8; ++r)
    rl[((size_t)b * P2 + mt * 16 + mb + r) * P2 + nt * 16 + n0] = acc[r];
}

// ---- top-4 per row (value-desc, earlier index wins ties, like lax.top_k) ----
__global__ void k_topk(const float* __restrict__ rl, int* __restrict__ idx) {
  int row = blockIdx.x * blockDim.x + threadIdx.x;
  if (row >= BB * P2) return;
  const float* rp = rl + (size_t)row * P2;
  float bv[TOPK]; int bi[TOPK];
#pragma unroll
  for (int j = 0; j < TOPK; ++j) { bv[j] = -1e30f; bi[j] = 0; }
  for (int n = 0; n < P2; ++n) {
    float val = rp[n];
    if (val > bv[TOPK - 1]) {
      int j = TOPK - 1;
      while (j > 0 && val > bv[j - 1]) { bv[j] = bv[j - 1]; bi[j] = bi[j - 1]; --j; }
      bv[j] = val; bi[j] = n;
    }
  }
#pragma unroll
  for (int j = 0; j < TOPK; ++j) idx[row * TOPK + j] = bi[j];
}

// ---- bi-level routing attention: one block per (b,window), 8 waves ----
// LDS: S 64KB + P 32KB + Ks 48KB = ~144KB (WGP has 320KB)
__global__ void __launch_bounds__(256)
k_attn(const f16* __restrict__ q, const f16* __restrict__ k,
       const f16* __restrict__ v, const int* __restrict__ idx,
       f16* __restrict__ o) {
  __shared__ alignas(16) float S[W2 * LKEY];     // 64x256 scores fp32
  __shared__ alignas(16) f16   P[W2 * LKEY];     // softmax probs fp16
  __shared__ alignas(16) f16   Ks[LKEY * CC];    // gathered K windows
  __shared__ float rmax[W2], rsum[W2];
  const float scale = rsqrtf((float)CC);     // C**-0.5
  int bp = blockIdx.x, b = bp / P2;
  int tid = threadIdx.x, wave = tid >> 5, lane = tid & 31;
  int sel[TOPK];
#pragma unroll
  for (int j = 0; j < TOPK; ++j) sel[j] = idx[bp * TOPK + j];

  // async-gather the 4 selected K windows into LDS: 256 rows x 192B
  {
    unsigned ldsK = (unsigned)(uintptr_t)(void*)&Ks[0];
#pragma unroll
    for (int it = 0; it < 12; ++it) {
      unsigned boff = (unsigned)(tid + it * 256) * 16u;   // 0..49136
      int row = (int)(boff / 192u);
      int win = (int)(boff % 192u);
      const char* g = (const char*)(k + ((size_t)(b * P2 + sel[row >> 6]) * W2
                                         + (row & 63)) * CC) + win;
      async_copy16(ldsK + boff, g);
    }
    async_wait0();
  }
  __syncthreads();

  for (int head = 0; head < NHEAD; ++head) {
    // --- S = scale * q_h @ k_h^T : 4 M-tiles x 16 N-tiles, K padded 12->32
#pragma unroll
    for (int i = 0; i < 8; ++i) {
      int tile = wave * 8 + i;
      int mt = tile >> 4, nt = tile & 15;
      const f16* qb = q + ((size_t)bp * W2 + mt * 16) * CC + head * HD;
      v8f acc = {};
      acc = wmma_f16(load_frag(qb, CC, 1, HD),
                     load_frag(&Ks[(nt * 16) * CC + head * HD], CC, 1, HD), acc);
      int n0 = lane & 15, mb = (lane & 16) ? 8 : 0;
#pragma unroll
      for (int r = 0; r < 8; ++r)
        S[(mt * 16 + mb + r) * LKEY + nt * 16 + n0] = acc[r] * scale;
    }
    __syncthreads();
    // --- row softmax stats
    if (tid < W2) {
      float mx = -1e30f;
      for (int n = 0; n < LKEY; ++n) mx = fmaxf(mx, S[tid * LKEY + n]);
      float sm = 0.f;
      for (int n = 0; n < LKEY; ++n) sm += __expf(S[tid * LKEY + n] - mx);
      rmax[tid] = mx; rsum[tid] = 1.0f / sm;
    }
    __syncthreads();
    for (int i2 = tid; i2 < W2 * LKEY; i2 += 256) {
      int rr = i2 >> 8;
      P[i2] = (f16)(__expf(S[i2] - rmax[rr]) * rsum[rr]);
    }
    __syncthreads();
    // --- O = P @ v_h : 4 M-tiles (waves 0-3), N padded 12->16, K=256
    if (wave < 4) {
      int mt = wave;
      v8f acc = {};
#pragma unroll
      for (int ks = 0; ks < 8; ++ks) {
        v16h a = load_frag(&P[(mt * 16) * LKEY + ks * 32], LKEY, 1, 32);
        const f16* vb = v + ((size_t)(b * P2 + sel[ks >> 1]) * W2 + (ks & 1) * 32) * CC
                          + head * HD;
        v16h bf = load_frag(vb, 1, CC, 32);   // B^T[n][k] = v[k][hd0+n]
        acc = wmma_f16(a, bf, acc);
      }
      int n0 = lane & 15, mb = (lane & 16) ? 8 : 0;
      if (n0 < HD) {
#pragma unroll
        for (int r = 0; r < 8; ++r)
          o[((size_t)bp * W2 + mt * 16 + mb + r) * CC + head * HD + n0] = (f16)acc[r];
      }
    }
    __syncthreads();
  }
}

// ---- LePE 5x5 depthwise on v + combine with o -> u fp16 raster token-major ----
__global__ void k_lepe(const f16* __restrict__ v, const f16* __restrict__ o,
                       const float* __restrict__ lw, const float* __restrict__ lb,
                       f16* __restrict__ u) {
  int m = blockIdx.x * blockDim.x + threadIdx.x;
  if (m >= NT) return;
  int b = m / HWN, t = m % HWN;
  int y = t / WWID, xx = t % WWID;
  int p  = (y >> 3) * (WWID >> 3) + (xx >> 3);
  int wi = (y & 7) * 8 + (xx & 7);
  size_t orow = ((size_t)(b * P2 + p) * W2 + wi) * CC;
  f16* up = u + (size_t)m * CC;
  for (int c = 0; c < CC; ++c) {
    float acc = lb[c];
    const float* wp = lw + c * 25;
    for (int dy = -2; dy <= 2; ++dy) {
      int yy = y + dy;
      if (yy < 0 || yy >= HH) continue;
      for (int dx = -2; dx <= 2; ++dx) {
        int x2 = xx + dx;
        if (x2 < 0 || x2 >= WWID) continue;
        int pp  = (yy >> 3) * (WWID >> 3) + (x2 >> 3);
        int wwi = (yy & 7) * 8 + (x2 & 7);
        acc += wp[(dy + 2) * 5 + (dx + 2)] *
               (float)v[((size_t)(b * P2 + pp) * W2 + wwi) * CC + c];
      }
    }
    up[c] = (f16)((float)o[orow + c] + acc);
  }
}

// ---- wo GEMM (staged) + bias + residual(x NCHW) -> x1 fp32 raster ----
__global__ void __launch_bounds__(256)
k_wo(const f16* __restrict__ u, const f16* __restrict__ w,
     const float* __restrict__ bias, const float* __restrict__ x,
     float* __restrict__ x1) {
  const size_t row0 = (size_t)blockIdx.x * MBLK;
  gemm_staged<CC>(u + row0 * CC, w, [&](int mw, int nt, v8f acc) {
    int lane = threadIdx.x & 31, n0 = lane & 15, mb = (lane & 16) ? 8 : 0;
    int n = nt * 16 + n0;
    float bv = bias[n];
#pragma unroll
    for (int r = 0; r < 8; ++r) {
      size_t m = row0 + mw * 16 + mb + r;
      int b = (int)(m / HWN), t = (int)(m % HWN);
      float xv = x[((size_t)b * CC + n) * HWN + t];
      x1[m * CC + n] = xv + acc[r] + bv;
    }
  });
}

// ---- LN2 over x1 -> hn fp16 ----
__global__ void k_ln2(const float* __restrict__ x1, const float* __restrict__ g,
                      const float* __restrict__ bta, f16* __restrict__ hn) {
  int m = blockIdx.x * blockDim.x + threadIdx.x;
  if (m >= NT) return;
  const float* xp = x1 + (size_t)m * CC;
  float mean = 0.f;
#pragma unroll 4
  for (int c = 0; c < CC; ++c) mean += xp[c];
  mean *= (1.0f / CC);
  float var = 0.f;
#pragma unroll 4
  for (int c = 0; c < CC; ++c) { float d = xp[c] - mean; var += d * d; }
  var *= (1.0f / CC);
  float rstd = rsqrtf(var + 1e-5f);
  f16* op = hn + (size_t)m * CC;
#pragma unroll 4
  for (int c = 0; c < CC; ++c) op[c] = (f16)((xp[c] - mean) * rstd * g[c] + bta[c]);
}

// ---- pin GEMM (staged): (NT x 96) @ (96 x 192) -> tb fp16 [token][192] ----
__global__ void __launch_bounds__(256)
k_pin(const f16* __restrict__ hn, const f16* __restrict__ w, f16* __restrict__ tb) {
  const size_t row0 = (size_t)blockIdx.x * MBLK;
  gemm_staged<2 * CC>(hn + row0 * CC, w, [&](int mw, int nt, v8f acc) {
    int lane = threadIdx.x & 31, n0 = lane & 15, mb = (lane & 16) ? 8 : 0;
#pragma unroll
    for (int r = 0; r < 8; ++r)
      tb[(row0 + mw * 16 + mb + r) * (2 * CC) + nt * 16 + n0] = (f16)acc[r];
  });
}

// ---- 3x3 depthwise on both halves + gated exact-GELU -> g fp16 [token][96] ----
__global__ void k_gate(const f16* __restrict__ tb, const float* __restrict__ dw,
                       f16* __restrict__ g) {
  int i = blockIdx.x * blockDim.x + threadIdx.x;
  if (i >= NT * CC) return;
  int m = i / CC, c = i % CC;
  int b = m / HWN, t = m % HWN;
  int y = t / WWID, xx = t % WWID;
  float a1 = 0.f, a2 = 0.f;
  const float* w1 = dw + c * 9;
  const float* w2 = dw + (CC + c) * 9;
  for (int dy = -1; dy <= 1; ++dy) {
    int yy = y + dy; if (yy < 0 || yy >= HH) continue;
    for (int dx = -1; dx <= 1; ++dx) {
      int x2 = xx + dx; if (x2 < 0 || x2 >= WWID) continue;
      size_t base = ((size_t)b * HWN + yy * WWID + x2) * (2 * CC);
      int wi = (dy + 1) * 3 + dx + 1;
      a1 += w1[wi] * (float)tb[base + c];
      a2 += w2[wi] * (float)tb[base + CC + c];
    }
  }
  float ge = 0.5f * a1 * (1.0f + erff(a1 * 0.70710678118654752f));
  g[(size_t)m * CC + c] = (f16)(ge * a2);
}

// ---- pout GEMM (staged) + residual(x1) -> out NCHW fp32 ----
__global__ void __launch_bounds__(256)
k_pout(const f16* __restrict__ g, const f16* __restrict__ w,
       const float* __restrict__ x1, float* __restrict__ out) {
  const size_t row0 = (size_t)blockIdx.x * MBLK;
  gemm_staged<CC>(g + row0 * CC, w, [&](int mw, int nt, v8f acc) {
    int lane = threadIdx.x & 31, n0 = lane & 15, mb = (lane & 16) ? 8 : 0;
    int n = nt * 16 + n0;
#pragma unroll
    for (int r = 0; r < 8; ++r) {
      size_t m = row0 + mw * 16 + mb + r;
      int b = (int)(m / HWN), t = (int)(m % HWN);
      out[((size_t)b * CC + n) * HWN + t] = x1[m * CC + n] + acc[r];
    }
  });
}

extern "C" void kernel_launch(void* const* d_in, const int* in_sizes, int n_in,
                              void* d_out, int out_size, void* d_ws, size_t ws_size,
                              hipStream_t stream) {
  const float* x     = (const float*)d_in[0];
  const float* ln1g  = (const float*)d_in[1];
  const float* ln1b  = (const float*)d_in[2];
  const float* qkvw  = (const float*)d_in[3];
  const float* qkvb  = (const float*)d_in[4];
  const float* wow   = (const float*)d_in[5];
  const float* wob   = (const float*)d_in[6];
  const float* lepew = (const float*)d_in[7];
  const float* lepeb = (const float*)d_in[8];
  const float* ln2g  = (const float*)d_in[9];
  const float* ln2b  = (const float*)d_in[10];
  const float* pinw  = (const float*)d_in[11];
  const float* dww   = (const float*)d_in[12];
  const float* poutw = (const float*)d_in[13];

  char* ws = (char*)d_ws;
  size_t off = 0;
  auto alloc = [&](size_t bytes) -> void* {
    void* p = ws + off;
    off = (off + bytes + 255) & ~(size_t)255;
    return p;
  };
  const size_t sz_tok16 = (size_t)NT * CC * sizeof(f16);
  f16*   xw     = (f16*)alloc(sz_tok16);
  f16*   q      = (f16*)alloc(sz_tok16);
  f16*   k      = (f16*)alloc(sz_tok16);
  f16*   v      = (f16*)alloc(sz_tok16 + 1024);     // pad for N-padded AV frag reads
  f16*   o      = (f16*)alloc(sz_tok16);
  f16*   u      = (f16*)alloc(sz_tok16);
  float* x1     = (float*)alloc((size_t)NT * CC * sizeof(float));
  f16*   hn     = (f16*)alloc(sz_tok16);
  f16*   tb     = (f16*)alloc((size_t)NT * 2 * CC * sizeof(f16));
  f16*   gbuf   = (f16*)alloc(sz_tok16);
  f16*   qw16   = (f16*)alloc((size_t)3 * CC * CC * sizeof(f16));
  f16*   wo16   = (f16*)alloc((size_t)CC * CC * sizeof(f16));
  f16*   pin16  = (f16*)alloc((size_t)2 * CC * CC * sizeof(f16));
  f16*   pout16 = (f16*)alloc((size_t)CC * CC * sizeof(f16));
  f16*   qm     = (f16*)alloc((size_t)BB * P2 * CC * sizeof(f16));
  f16*   km     = (f16*)alloc((size_t)BB * P2 * CC * sizeof(f16));
  float* rl     = (float*)alloc((size_t)BB * P2 * P2 * sizeof(float));
  int*   idx    = (int*)alloc((size_t)BB * P2 * TOPK * sizeof(int));

  // weight conversions
  k_cvt16<<<(3 * CC * CC + 255) / 256, 256, 0, stream>>>(qkvw, qw16, 3 * CC * CC);
  k_cvt16<<<(CC * CC + 255) / 256, 256, 0, stream>>>(wow, wo16, CC * CC);
  k_cvt16<<<(2 * CC * CC + 255) / 256, 256, 0, stream>>>(pinw, pin16, 2 * CC * CC);
  k_cvt16<<<(CC * CC + 255) / 256, 256, 0, stream>>>(poutw, pout16, CC * CC);

  k_ln1_win<<<NT / 256, 256, 0, stream>>>(x, ln1g, ln1b, xw);
  k_qkv<<<NT / MBLK, 256, 0, stream>>>(xw, qw16, qkvb, q, k, v);
  k_means<<<(BB * P2 * CC + 255) / 256, 256, 0, stream>>>(q, k, qm, km);
  k_route<<<(BB * 36 * 36 + 7) / 8, 256, 0, stream>>>(qm, km, rl);
  k_topk<<<(BB * P2 + 255) / 256, 256, 0, stream>>>(rl, idx);
  k_attn<<<BB * P2, 256, 0, stream>>>(q, k, v, idx, o);
  k_lepe<<<NT / 256, 256, 0, stream>>>(v, o, lepew, lepeb, u);
  k_wo<<<NT / MBLK, 256, 0, stream>>>(u, wo16, wob, x, x1);
  k_ln2<<<NT / 256, 256, 0, stream>>>(x1, ln2g, ln2b, hn);
  k_pin<<<NT / MBLK, 256, 0, stream>>>(hn, pin16, tb);
  k_gate<<<(NT * CC + 255) / 256, 256, 0, stream>>>(tb, dww, gbuf);
  k_pout<<<NT / MBLK, 256, 0, stream>>>(gbuf, pout16, x1, (float*)d_out);
}
